// ResonanceAttention_45818711114238
// MI455X (gfx1250) — matrix operand order repaired and checked by reading, not compiled, not run on hardware
//
#include <hip/hip_runtime.h>
#include <hip/hip_bf16.h>
#include <math.h>

// ---------------------------------------------------------------------------
// Types for WMMA fragments (wave32, 16x16x32 f16 -> f32)
// ---------------------------------------------------------------------------
typedef __attribute__((ext_vector_type(16))) _Float16 v16h;
typedef __attribute__((ext_vector_type(8)))  _Float16 v8h;
typedef __attribute__((ext_vector_type(8)))  float    v8f;
typedef __attribute__((ext_vector_type(4)))  unsigned u32x4;
typedef __attribute__((ext_vector_type(4)))  int      i32x4;
typedef __attribute__((ext_vector_type(8)))  int      i32x8;

__device__ __forceinline__ v8f wmma_f16f32(v16h a, v16h b, v8f c) {
  // (neg_a, A, neg_b, B, c_mod, C, reuse_a, reuse_b)
  return __builtin_amdgcn_wmma_f32_16x16x32_f16(false, a, false, b, (short)0, c,
                                                false, false);
}

// A-matrix fragment (16x32 f16): lane holds row = lane%16,
// grp = lane/16; elements 0..15 -> K = grp*16+e
__device__ __forceinline__ v16h frag_a16(const _Float16* rowBase, int grp) {
  v8h lo = *(const v8h*)(rowBase + grp * 16);
  v8h hi = *(const v8h*)(rowBase + grp * 16 + 8);
  return __builtin_shufflevector(lo, hi, 0, 1, 2, 3, 4, 5, 6, 7,
                                 8, 9, 10, 11, 12, 13, 14, 15);
}
// B fragment: lane col = lane%16, grp = lane/16; 16 contiguous K values.
__device__ __forceinline__ v16h frag_b16(const _Float16* p) {
  v8h lo = *(const v8h*)p;
  v8h hi = *(const v8h*)(p + 8);
  return __builtin_shufflevector(lo, hi, 0, 1, 2, 3, 4, 5, 6, 7,
                                 8, 9, 10, 11, 12, 13, 14, 15);
}

// ---------------------------------------------------------------------------
// CDNA5 async global -> LDS copy (ASYNCcnt tracked), 16 bytes per lane.
// ---------------------------------------------------------------------------
__device__ __forceinline__ void async_cp16(unsigned ldsOff, const void* gptr) {
  asm volatile("global_load_async_to_lds_b128 %0, %1, off"
               :: "v"(ldsOff), "v"(gptr) : "memory");
}
template <int N>
__device__ __forceinline__ void async_wait() {
  asm volatile("s_wait_asynccnt %0" :: "n"(N) : "memory");
}
template <typename T>
__device__ __forceinline__ unsigned lds_off(const T* p) {
  return (unsigned)(unsigned long long)p;  // low 32 bits = wave-relative LDS addr
}

// ---------------------------------------------------------------------------
// Tensor Data Mover: 2D f16 tile (rows x cols, row stride in elements) -> LDS,
// with optional LDS row padding via the D# pad fields.
// D# packing per CDNA5 ISA.
// ---------------------------------------------------------------------------
#if __has_builtin(__builtin_amdgcn_tensor_load_to_lds)
#define HAVE_TDM 1
__device__ __forceinline__ void tdm_load_2d_f16(unsigned lds, const void* gaddr,
                                                unsigned rows, unsigned cols,
                                                unsigned rowStride,
                                                unsigned tensorRows,
                                                int padEnable, int padInterval,
                                                int padAmount) {
  unsigned long long ga = (unsigned long long)gaddr;
  u32x4 g0;
  g0.x = 1u;                                       // count=1, user descriptor
  g0.y = lds;                                      // lds_addr
  g0.z = (unsigned)(ga & 0xffffffffu);             // global_addr[31:0]
  g0.w = (unsigned)((ga >> 32) & 0x01ffffffu) | (2u << 30);  // [56:32] | type=2
  i32x8 g1;
  g1[0] = (1 << 16)                                // data_size=1 (2 bytes)
        | (padEnable << 20)                        // pad_enable
        | (padInterval << 22)                      // pad_interval (code)
        | (padAmount << 25);                       // pad_amount (code)
  g1[1] = (int)(cols << 16);       // atomic_addr=0 | tensor_dim0[15:0]
  g1[2] = (int)(tensorRows << 16); // tensor_dim0[31:16]=0 | tensor_dim1[15:0]
  g1[3] = (int)(cols << 16);       // tensor_dim1[31:16]=0 | tile_dim0
  g1[4] = (int)rows;               // tile_dim1 | tile_dim2=0
  g1[5] = (int)rowStride;          // tensor_dim0_stride[31:0]
  g1[6] = 0;
  g1[7] = 0;
  i32x4 z4 = {0, 0, 0, 0};
#if defined(__clang_major__) && __clang_major__ >= 23
  i32x8 z8 = {0, 0, 0, 0, 0, 0, 0, 0};
  __builtin_amdgcn_tensor_load_to_lds(g0, g1, z4, z4, z8, 0);
#else
  __builtin_amdgcn_tensor_load_to_lds(g0, g1, z4, z4, 0);
#endif
}
__device__ __forceinline__ void tdm_wait0() {
  __builtin_amdgcn_s_wait_tensorcnt(0);
}
#endif

// ---------------------------------------------------------------------------
// f32 -> f16 convert
// ---------------------------------------------------------------------------
__global__ __launch_bounds__(256) void f32_to_f16_kernel(const float* __restrict__ s,
                                                         _Float16* __restrict__ d,
                                                         int n) {
  int i = blockIdx.x * 256 + threadIdx.x;
  if (i < n) d[i] = (_Float16)s[i];
}

// ---------------------------------------------------------------------------
// GEMM: C[M,N] = A[M,K] @ W[N,K]^T + bias (torch Linear), f16 in, f32 acc.
// WG = 256 threads = 8 waves (4 x 2), tile 128x128, BK = 32, double-buffered
// async global->LDS staging (prefetch k+1 while computing k).
// Each wave: 32x64 = 2x4 WMMA tiles -> 8 v_wmma per k-step.
// ---------------------------------------------------------------------------
template <int OUT_F16>
__global__ __launch_bounds__(256) void gemm_nt_kernel(
    const _Float16* __restrict__ A, const _Float16* __restrict__ W,
    const float* __restrict__ bias, void* __restrict__ C, int M, int N, int K) {
  __shared__ _Float16 At[2][128 * 32];  // 2 x 8 KB
  __shared__ _Float16 Bt[2][128 * 32];  // 2 x 8 KB

  const int tid = threadIdx.x;
  const int wave = tid >> 5, lane = tid & 31;
  const int laneRow = lane & 15, laneGrp = lane >> 4;
  const int m0 = blockIdx.y * 128, n0 = blockIdx.x * 128;
  const int wrow = wave >> 1, wcol = wave & 1;  // 4x2 waves: 32 rows x 64 cols

  v8f acc[2][4] = {};

  // 4 async instructions per stage call (ASYNCcnt += 4 per wave)
  auto stage = [&](int buf, int k0) {
#pragma unroll
    for (int i = 0; i < 2; ++i) {
      int c = tid + i * 256;  // 512 x 16B chunks per tile
      int r = c >> 2, s = c & 3;
      async_cp16(lds_off(&At[buf][r * 32 + s * 8]),
                 A + (size_t)(m0 + r) * K + k0 + s * 8);
      async_cp16(lds_off(&Bt[buf][r * 32 + s * 8]),
                 W + (size_t)(n0 + r) * K + k0 + s * 8);
    }
  };

  const int steps = K / 32;
  stage(0, 0);
#pragma unroll 2
  for (int ks = 0; ks < steps; ++ks) {
    const int buf = ks & 1;
    if (ks + 1 < steps) {
      stage(buf ^ 1, (ks + 1) * 32);  // prefetch next tile
      async_wait<4>();                // drain only the older 4 copies
    } else {
      async_wait<0>();
    }
    __syncthreads();

    v16h af[2], bf[4];
#pragma unroll
    for (int mt = 0; mt < 2; ++mt)
      af[mt] = frag_a16(&At[buf][(wrow * 32 + mt * 16 + laneRow) * 32], laneGrp);
#pragma unroll
    for (int nt = 0; nt < 4; ++nt)
      bf[nt] = frag_b16(&Bt[buf][(wcol * 64 + nt * 16 + laneRow) * 32 + laneGrp * 16]);
#pragma unroll
    for (int mt = 0; mt < 2; ++mt)
#pragma unroll
      for (int nt = 0; nt < 4; ++nt)
        acc[mt][nt] = wmma_f16f32(af[mt], bf[nt], acc[mt][nt]);
    __syncthreads();  // all waves done with buf before it is re-staged
  }

  // Epilogue: C/D layout = lane col = lane%16, row = vgpr + 8*(lane/16)
#pragma unroll
  for (int mt = 0; mt < 2; ++mt)
#pragma unroll
    for (int nt = 0; nt < 4; ++nt) {
      int n = n0 + wcol * 64 + nt * 16 + laneRow;
      float bv = bias ? bias[n] : 0.0f;
#pragma unroll
      for (int v = 0; v < 8; ++v) {
        int m = m0 + wrow * 32 + mt * 16 + v + 8 * laneGrp;
        float val = acc[mt][nt][v] + bv;
        if (OUT_F16)
          ((_Float16*)C)[(size_t)m * N + n] = (_Float16)val;
        else
          ((float*)C)[(size_t)m * N + n] = val;
      }
    }
}

// ---------------------------------------------------------------------------
// Fused single-pass attention exploiting CDNA5's 320 KB LDS per WGP:
// the whole 32x1024 f32 score strip and 32x1024 f16 probability strip stay
// resident in (dynamic) LDS, so QK^T and exp() are computed exactly once.
// One WG per (b, h, 32-query block); hd = 64.
//   Phase 1: QK^T via WMMA into sFull (kT double-buffered async staging)
//   Phase 2: full-row softmax + entropy, probabilities -> pFull (f16)
//   Phase 3: P @ V via WMMA (V staged transposed per 128-key chunk)
// ---------------------------------------------------------------------------
#define SQ 32             // queries per workgroup
#define CK 128            // keys per chunk (8 waves x 16 keys)
#define VSTRIDE (CK + 8)  // padded V^T row stride (breaks bank conflicts)
#define ATT_S 1024
#define PSTRIDE (ATT_S + 8)  // padded probability row stride
#define QSTRIDE 66           // padded Q row stride (64 + 1 DWORD)

// dynamic LDS layout (bytes)
#define OFF_SFULL 0
#define SZ_SFULL  (SQ * ATT_S * 4)                       // 131072
#define OFF_PFULL (OFF_SFULL + SZ_SFULL)
#define SZ_PFULL  (SQ * PSTRIDE * 2)                     // 66048
#define OFF_KT    (OFF_PFULL + SZ_PFULL)
#define SZ_KT     (2 * CK * 64 * 2)                      // 32768
#define OFF_VTT   (OFF_KT + SZ_KT)
#define SZ_VTT    (64 * VSTRIDE * 2)                     // 17408
#define OFF_QT    (OFF_VTT + SZ_VTT)
#define SZ_QT     (SQ * QSTRIDE * 2)                     // 4224
#define ATT_SMEM  (OFF_QT + SZ_QT)                       // ~246 KB < 320 KB

__global__ __launch_bounds__(256) void attn_kernel(
    const _Float16* __restrict__ Q, const _Float16* __restrict__ K,
    const _Float16* __restrict__ V, const float* __restrict__ res_bias,
    _Float16* __restrict__ O, float* __restrict__ entAcc, int B, int S, int D,
    int H) {
  extern __shared__ __align__(16) char smem[];
  float*    sFull = (float*)(smem + OFF_SFULL);
  _Float16* pFull = (_Float16*)(smem + OFF_PFULL);
  _Float16* kT    = (_Float16*)(smem + OFF_KT);
  _Float16* vTt   = (_Float16*)(smem + OFF_VTT);
  _Float16* qT    = (_Float16*)(smem + OFF_QT);

  const int tid = threadIdx.x;
  const int wave = tid >> 5, lane = tid & 31;
  const int laneRow = lane & 15, laneGrp = lane >> 4;
  const int bh = blockIdx.y;
  const int b = bh / H, h = bh % H;
  const int q0 = blockIdx.x * SQ;
  const _Float16* Qb = Q + (size_t)(b * S + q0) * D + h * 64;
  const _Float16* Kb = K + (size_t)b * S * D + h * 64;
  const _Float16* Vb = V + (size_t)b * S * D + h * 64;
  const float rb = res_bias[h];
  const float scale = 0.125f;  // 1/sqrt(64)
  const int nChunks = S / CK;

  // ---- Stage Q block (32 x 64 f16, row stride D), LDS rows padded ----
#ifdef HAVE_TDM
  if (wave == 0) {
    // pad_interval code 4 => every 32 DWORDs (one 128B row), pad_amount code 0
    // => 1 DWORD (2 halves): LDS row stride becomes QSTRIDE = 66 halves.
    tdm_load_2d_f16(lds_off(&qT[0]), Qb, SQ, 64, (unsigned)D, (unsigned)S,
                    /*padEnable=*/1, /*padInterval=*/4, /*padAmount=*/0);
    tdm_wait0();
  }
#else
  {
    int r = tid >> 3, s = tid & 7;
    async_cp16(lds_off(&qT[r * QSTRIDE + s * 8]), Qb + (size_t)r * D + s * 8);
  }
  async_wait<0>();
#endif
  __syncthreads();

  // Q fragments held in registers for the whole kernel
  v16h qf[2][2];
#pragma unroll
  for (int mt = 0; mt < 2; ++mt)
#pragma unroll
    for (int g2 = 0; g2 < 2; ++g2)
      qf[mt][g2] = frag_a16(&qT[(mt * 16 + laneRow) * QSTRIDE + g2 * 32], laneGrp);

  // 4 async instructions per stage call (ASYNCcnt += 4 per wave)
  auto stage_k = [&](int buf, int c) {
#pragma unroll
    for (int i = 0; i < 4; ++i) {
      int idx = tid + i * 256;  // 1024 x 16B chunks
      int r = idx >> 3, s = idx & 7;
      async_cp16(lds_off(&kT[buf * CK * 64 + r * 64 + s * 8]),
                 Kb + (size_t)(c * CK + r) * D + s * 8);
    }
  };

  // -------- Phase 1: all scores into sFull (double-buffered K tiles) -------
  stage_k(0, 0);
  for (int c = 0; c < nChunks; ++c) {
    const int buf = c & 1;
    if (c + 1 < nChunks) {
      stage_k(buf ^ 1, c + 1);  // prefetch next K chunk
      async_wait<4>();          // drain only the older 4 copies
    } else {
      async_wait<0>();
    }
    __syncthreads();
    const _Float16* kb = &kT[buf * CK * 64];
    v16h bf0 = frag_b16(&kb[(wave * 16 + laneRow) * 64 + 0 + laneGrp * 16]);
    v16h bf1 = frag_b16(&kb[(wave * 16 + laneRow) * 64 + 32 + laneGrp * 16]);
#pragma unroll
    for (int mt = 0; mt < 2; ++mt) {
      v8f acc = {};
      acc = wmma_f16f32(qf[mt][0], bf0, acc);
      acc = wmma_f16f32(qf[mt][1], bf1, acc);
#pragma unroll
      for (int v = 0; v < 8; ++v) {
        int row = mt * 16 + v + 8 * laneGrp;
        sFull[row * ATT_S + c * CK + wave * 16 + laneRow] = acc[v] * scale + rb;
      }
    }
    __syncthreads();  // protect kT[buf] before it is re-staged
  }

  // -------- Phase 2: full-row softmax + entropy, p -> pFull (f16) ----------
  float entLane[4] = {0.0f, 0.0f, 0.0f, 0.0f};
#pragma unroll
  for (int rr = 0; rr < 4; ++rr) {
    const int r = wave * 4 + rr;  // each wave owns 4 rows exclusively
    const float* srow = &sFull[r * ATT_S];
    _Float16* prow = &pFull[r * PSTRIDE];
    float m = -INFINITY;
    for (int i = 0; i < ATT_S / 32; ++i)
      m = fmaxf(m, srow[lane + 32 * i]);
    for (int off = 16; off; off >>= 1) m = fmaxf(m, __shfl_xor(m, off, 32));
    float l = 0.0f;
    for (int i = 0; i < ATT_S / 32; ++i) {
      float e = __expf(srow[lane + 32 * i] - m);
      l += e;
      prow[lane + 32 * i] = (_Float16)e;  // unnormalized, one exp per element
    }
    for (int off = 16; off; off >>= 1) l += __shfl_xor(l, off, 32);
    const float invl = 1.0f / l;
    for (int i = 0; i < ATT_S / 32; ++i) {
      float p = (float)prow[lane + 32 * i] * invl;
      prow[lane + 32 * i] = (_Float16)p;
      float pe = p + 1e-10f;
      entLane[rr] += pe * __logf(pe);
    }
  }
  __syncthreads();  // pFull complete before P@V

  // -------- Phase 3: P[32 x S] @ V[S x 64] ---------------------------------
  const int mtO = wave >> 2, ntO = wave & 3;  // this wave's 16x16 output tile
  v8f oacc = {};
  for (int c = 0; c < nChunks; ++c) {
    if (c + 1 < nChunks)  // hint the next V chunk toward the caches
      __builtin_prefetch(Vb + (size_t)((c + 1) * CK + (tid >> 3)) * D, 0, 0);
    // Stage V transposed: load v8h along hd, scatter to vTt[hd][key]
#pragma unroll
    for (int i = 0; i < 4; ++i) {
      int idx = tid + i * 256;
      int r = idx >> 3, s = idx & 7;  // key row r, hd segment s
      v8h vv = *(const v8h*)(Vb + (size_t)(c * CK + r) * D + s * 8);
#pragma unroll
      for (int j = 0; j < 8; ++j) vTt[(s * 8 + j) * VSTRIDE + r] = vv[j];
    }
    __syncthreads();
#pragma unroll
    for (int kk = 0; kk < 4; ++kk) {
      v16h pa = frag_a16(&pFull[(mtO * 16 + laneRow) * PSTRIDE + c * CK + kk * 32],
                         laneGrp);
      v16h vb = frag_b16(&vTt[(ntO * 16 + laneRow) * VSTRIDE + kk * 32 + laneGrp * 16]);
      oacc = wmma_f16f32(pa, vb, oacc);
    }
    __syncthreads();  // protect vTt before next chunk overwrites it
  }

  // Store output tile (f16 attn_out)
#pragma unroll
  for (int v = 0; v < 8; ++v) {
    int row = mtO * 16 + v + 8 * laneGrp;
    int col = ntO * 16 + laneRow;
    O[(size_t)(b * S + q0 + row) * D + h * 64 + col] = (_Float16)oacc[v];
  }
  // Entropy: reduce per-row partials over the wave, accumulate -sum
#pragma unroll
  for (int rr = 0; rr < 4; ++rr) {
    float s = entLane[rr];
    for (int off = 16; off; off >>= 1) s += __shfl_xor(s, off, 32);
    if (lane == 0) atomicAdd(&entAcc[bh], -s);
  }
}

// ---------------------------------------------------------------------------
// Finalize: ent[b][h] = entAcc / S ; iters = 1 if all batch-mean ent < 3 else 4
// (the resonance loop is a fixed point: attn and ent never change)
// ---------------------------------------------------------------------------
__global__ void finalize_kernel(const float* __restrict__ entAcc,
                                float* __restrict__ outTail, int B, int H,
                                int S) {
  __shared__ float e[64];
  int t = threadIdx.x;
  if (t < B * H) {
    float v = entAcc[t] / (float)S;
    e[t] = v;
    outTail[t] = v;
  }
  __syncthreads();
  if (t == 0) {
    bool done = true;
    for (int b = 0; b < B; ++b) {
      float m = 0.0f;
      for (int h = 0; h < H; ++h) m += e[b * H + h];
      m /= (float)H;
      if (!(m < 3.0f)) done = false;
    }
    outTail[B * H] = done ? 1.0f : 4.0f;  // iters
  }
}

// ---------------------------------------------------------------------------
// Host launcher
// ---------------------------------------------------------------------------
extern "C" void kernel_launch(void* const* d_in, const int* in_sizes, int n_in,
                              void* d_out, int out_size, void* d_ws,
                              size_t ws_size, hipStream_t stream) {
  (void)in_sizes; (void)n_in; (void)out_size; (void)ws_size;
  const int B = 2, S = 1024, D = 1024, H = 16;
  const int M = B * S;

  const float* x  = (const float*)d_in[0];
  const float* wq = (const float*)d_in[1];
  const float* bq = (const float*)d_in[2];
  const float* wk = (const float*)d_in[3];
  const float* bk = (const float*)d_in[4];
  const float* wv = (const float*)d_in[5];
  const float* bv = (const float*)d_in[6];
  const float* wo = (const float*)d_in[7];
  const float* bo = (const float*)d_in[8];
  const float* rb = (const float*)d_in[9];

  char* p = (char*)d_ws;
  _Float16* xh  = (_Float16*)p; p += (size_t)M * D * 2;
  _Float16* wqh = (_Float16*)p; p += (size_t)D * D * 2;
  _Float16* wkh = (_Float16*)p; p += (size_t)D * D * 2;
  _Float16* wvh = (_Float16*)p; p += (size_t)D * D * 2;
  _Float16* woh = (_Float16*)p; p += (size_t)D * D * 2;
  _Float16* Qh  = (_Float16*)p; p += (size_t)M * D * 2;
  _Float16* Kh  = (_Float16*)p; p += (size_t)M * D * 2;
  _Float16* Vh  = (_Float16*)p; p += (size_t)M * D * 2;
  _Float16* Oh  = (_Float16*)p; p += (size_t)M * D * 2;
  float* entAcc = (float*)p;

  hipMemsetAsync(entAcc, 0, (size_t)B * H * sizeof(float), stream);

  int n = M * D;
  f32_to_f16_kernel<<<(n + 255) / 256, 256, 0, stream>>>(x, xh, n);
  n = D * D;
  f32_to_f16_kernel<<<(n + 255) / 256, 256, 0, stream>>>(wq, wqh, n);
  f32_to_f16_kernel<<<(n + 255) / 256, 256, 0, stream>>>(wk, wkh, n);
  f32_to_f16_kernel<<<(n + 255) / 256, 256, 0, stream>>>(wv, wvh, n);
  f32_to_f16_kernel<<<(n + 255) / 256, 256, 0, stream>>>(wo, woh, n);

  dim3 gg(D / 128, M / 128);
  gemm_nt_kernel<1><<<gg, 256, 0, stream>>>(xh, wqh, bq, Qh, M, D, D);
  gemm_nt_kernel<1><<<gg, 256, 0, stream>>>(xh, wkh, bk, Kh, M, D, D);
  gemm_nt_kernel<1><<<gg, 256, 0, stream>>>(xh, wvh, bv, Vh, M, D, D);

  // Single-pass attention: ~246 KB dynamic LDS (CDNA5 WGP has 320 KB)
  attn_kernel<<<dim3(S / SQ, B * H), 256, ATT_SMEM, stream>>>(
      Qh, Kh, Vh, rb, Oh, entAcc, B, S, D, H);

  gemm_nt_kernel<0><<<gg, 256, 0, stream>>>(Oh, woh, bo, d_out, M, D, D);

  finalize_kernel<<<1, 64, 0, stream>>>(entAcc, (float*)d_out + (size_t)M * D,
                                        B, H, S);
}
